// MotionModelCont_79602923864471
// MI455X (gfx1250) — compile-verified, matrix operand order checked by reading, not run on hardware
//
#include <hip/hip_runtime.h>
#include <hip/hip_bf16.h>
#include <math.h>

// ---------------- model dimensions ----------------
#define Bc   8
#define LTc  128
#define Tc   196
#define Dc   263
#define Hc   1024
#define NHc  16
#define DHc  64
#define Fc   4096
#define NLc  4
#define Vc   32000
#define Sc   (LTc + Tc)          // 324
#define DPAD 288                 // D padded to multiple of 32 for WMMA K
#define MS   (Bc * Sc)           // 2592 rows through transformer
#define MT   (Bc * Tc)           // 1568 motion rows
#define MLM  (Bc * (LTc - 1))    // 1016 LM-head rows
#define LCH  4000                // vocab chunk for online log-softmax (8 chunks)

typedef __attribute__((ext_vector_type(16))) __bf16 v16bf;
typedef __attribute__((ext_vector_type(8)))  float  v8f;

// ---------------- CDNA5 async-to-LDS helpers ----------------
__device__ __forceinline__ void async_b128(unsigned lds_off, const void* gptr) {
  unsigned long long ga = (unsigned long long)(size_t)gptr;
  asm volatile("global_load_async_to_lds_b128 %0, %1, off"
               :: "v"(lds_off), "v"(ga) : "memory");
}
// wait until this wave's outstanding async ops <= 4 (previous tile done,
// next tile may still be in flight: async loads complete in issue order)
__device__ __forceinline__ void wait_async4() {
  asm volatile("s_wait_asynccnt 4" ::: "memory");
}
__device__ __forceinline__ void wait_async0() {
  asm volatile("s_wait_asynccnt 0" ::: "memory");
}

__device__ __forceinline__ float gelu_exact(float x) {
  return 0.5f * x * (1.0f + erff(x * 0.70710678118654752f));
}

// ---------------- WMMA GEMM: C[M,N] = A[M,K] * Btr[N,K]^T (+epilogue) ------
// op: 0 = none, 1 = +bias, 2 = +bias then gelu, 3 = +res (residual add)
// Double-buffered LDS tiles; async DMA for tile k+1 overlaps WMMA on tile k.
#define TMt 128
#define TNt 128
#define TKt 32
#define LDSU 20   // uints per LDS row: 16 data (32 bf16) + 4 pad -> 80B stride

union FragBF { unsigned u[8]; v16bf v; };
union FragF  { float f[8];    v8f   v; };

__global__ __launch_bounds__(256)
void k_gemm(const __bf16* __restrict__ A, const __bf16* __restrict__ Btr,
            float* __restrict__ C, const float* __restrict__ bias,
            const float* __restrict__ res, int M, int N, int K, int op) {
  __shared__ unsigned sA[2][TMt * LDSU];
  __shared__ unsigned sB[2][TNt * LDSU];

  const int tid  = threadIdx.x;
  const int lane = tid & 31;
  const int wave = tid >> 5;      // 0..7
  const int wm   = wave & 3;      // 4 waves along M, 32 rows each
  const int wn   = wave >> 2;     // 2 waves along N, 64 cols each
  const int m0   = blockIdx.x * TMt;
  const int n0   = blockIdx.y * TNt;
  const int hi   = lane >> 4;
  const int lo   = lane & 15;

  FragF acc[2][4];
  #pragma unroll
  for (int i = 0; i < 2; ++i)
    #pragma unroll
    for (int j = 0; j < 4; ++j)
      #pragma unroll
      for (int r = 0; r < 8; ++r) acc[i][j].f[r] = 0.0f;

  // stage one 128x32 A tile + 128x32 Btr tile into LDS buffer bsel
  // (4 async b128 ops per thread per tile)
  auto stage = [&](int kt, int bsel) {
    int k0 = kt * TKt;
    #pragma unroll
    for (int t = 0; t < 2; ++t) {
      int ch  = tid + t * 256;    // 0..511 chunks (row, 16B piece)
      int row = ch >> 2;
      int c   = ch & 3;
      int ar = m0 + row; if (ar >= M) ar = M - 1;
      async_b128((unsigned)(size_t)&sA[bsel][row * LDSU + c * 4],
                 A + (size_t)ar * K + k0 + c * 8);
      int br = n0 + row; if (br >= N) br = N - 1;
      async_b128((unsigned)(size_t)&sB[bsel][row * LDSU + c * 4],
                 Btr + (size_t)br * K + k0 + c * 8);
    }
  };

  const int nk = K / TKt;
  stage(0, 0);
  for (int kt = 0; kt < nk; ++kt) {
    const int cur = kt & 1;
    const bool more = (kt + 1 < nk);
    if (more) stage(kt + 1, cur ^ 1);     // overlap next-tile DMA with compute
    if (more) wait_async4(); else wait_async0();
    __syncthreads();                      // current tile visible to all waves

    FragBF af[2], bf[4];
    #pragma unroll
    for (int i = 0; i < 2; ++i) {
      const unsigned* rp = &sA[cur][(wm * 32 + i * 16 + lo) * LDSU];
      #pragma unroll
      for (int d = 0; d < 4; ++d) {
        af[i].u[d]     = rp[hi * 4 + d];
        af[i].u[4 + d] = rp[8 + hi * 4 + d];
      }
    }
    #pragma unroll
    for (int j = 0; j < 4; ++j) {
      const unsigned* rp = &sB[cur][(wn * 64 + j * 16 + lo) * LDSU];
      #pragma unroll
      for (int d = 0; d < 4; ++d) {
        bf[j].u[d]     = rp[hi * 4 + d];
        bf[j].u[4 + d] = rp[8 + hi * 4 + d];
      }
    }
    #pragma unroll
    for (int i = 0; i < 2; ++i)
      #pragma unroll
      for (int j = 0; j < 4; ++j)
        acc[i][j].v = __builtin_amdgcn_wmma_f32_16x16x32_bf16(
            false, af[i].v, false, bf[j].v, (short)0, acc[i][j].v, false, false);
    __syncthreads();                      // safe to overwrite 'cur' next round
  }

  // ---- epilogue: C layout row = m0+wm*32+i*16+hi*8+r, col = n0+wn*64+j*16+lo
  #pragma unroll
  for (int i = 0; i < 2; ++i)
    #pragma unroll
    for (int j = 0; j < 4; ++j) {
      int col = n0 + wn * 64 + j * 16 + lo;
      #pragma unroll
      for (int r = 0; r < 8; ++r) {
        int rowm = m0 + wm * 32 + i * 16 + hi * 8 + r;
        if (rowm < M && col < N) {
          float vv = acc[i][j].f[r];
          if (op == 1 || op == 2) vv += bias[col];
          if (op == 2) vv = gelu_exact(vv);
          if (op == 3) vv += res[(size_t)rowm * N + col];
          C[(size_t)rowm * N + col] = vv;
        }
      }
    }
}

// ---------------- conversion / transpose kernels ----------------
__global__ void k_cvt_tr(const float* __restrict__ src, __bf16* __restrict__ dst,
                         int K, int N, int Kpad) {
  size_t i = (size_t)blockIdx.x * 256 + threadIdx.x;
  if (i >= (size_t)N * Kpad) return;
  int n = (int)(i / Kpad), kp = (int)(i % Kpad);
  dst[i] = (kp < K) ? (__bf16)src[(size_t)kp * N + n] : (__bf16)0.0f;
}

__global__ void k_cvt_pad(const float* __restrict__ src, __bf16* __restrict__ dst,
                          int rows, int K, int Kpad) {
  size_t i = (size_t)blockIdx.x * 256 + threadIdx.x;
  if (i >= (size_t)rows * Kpad) return;
  int kp = (int)(i % Kpad); size_t row = i / Kpad;
  dst[i] = (kp < K) ? (__bf16)src[row * K + kp] : (__bf16)0.0f;
}

// ---------------- norms ----------------
__global__ void k_rmsnorm(const float* __restrict__ x, const float* __restrict__ g,
                          __bf16* __restrict__ out, int rows) {
  int row = blockIdx.x; if (row >= rows) return;
  __shared__ float red[256];
  int tid = threadIdx.x;
  const float* xr = x + (size_t)row * Hc;
  float s = 0.f;
  for (int i = tid; i < Hc; i += 256) { float v = xr[i]; s += v * v; }
  red[tid] = s; __syncthreads();
  for (int st = 128; st > 0; st >>= 1) { if (tid < st) red[tid] += red[tid + st]; __syncthreads(); }
  float rs = rsqrtf(red[0] / (float)Hc + 1e-6f);
  for (int i = tid; i < Hc; i += 256)
    out[(size_t)row * Hc + i] = (__bf16)(xr[i] * rs * g[i]);
}

__global__ void k_layernorm(const float* __restrict__ x, const float* __restrict__ g,
                            const float* __restrict__ b, float* __restrict__ out, int rows) {
  int row = blockIdx.x; if (row >= rows) return;
  __shared__ float red[256];
  int tid = threadIdx.x;
  const float* xr = x + (size_t)row * Hc;
  float s = 0.f;
  for (int i = tid; i < Hc; i += 256) s += xr[i];
  red[tid] = s; __syncthreads();
  for (int st = 128; st > 0; st >>= 1) { if (tid < st) red[tid] += red[tid + st]; __syncthreads(); }
  float mu = red[0] / (float)Hc; __syncthreads();
  float v = 0.f;
  for (int i = tid; i < Hc; i += 256) { float d = xr[i] - mu; v += d * d; }
  red[tid] = v; __syncthreads();
  for (int st = 128; st > 0; st >>= 1) { if (tid < st) red[tid] += red[tid + st]; __syncthreads(); }
  float rs = rsqrtf(red[0] / (float)Hc + 1e-5f);
  for (int i = tid; i < Hc; i += 256)
    out[(size_t)row * Hc + i] = (xr[i] - mu) * rs * g[i] + b[i];
}

// ---------------- sequence assembly ----------------
__global__ void k_posmask(const int* __restrict__ ids, const float* __restrict__ mm,
                          float* __restrict__ pos, int* __restrict__ am) {
  int b = blockIdx.x;
  if (threadIdx.x != 0) return;
  int cum = 0, mx = 0;
  for (int s = 0; s < LTc; ++s) {
    int t = (ids[b * LTc + s] != 0) ? 1 : 0;
    cum += t;
    int tp = cum - 1; if (tp < 0) tp = 0;
    pos[b * Sc + s] = (float)tp; am[b * Sc + s] = t;
    if (tp > mx) mx = tp;
  }
  int sp = mx + 1;
  pos[b * Sc + LTc] = (float)sp; am[b * Sc + LTc] = 1;
  for (int j = 1; j < Tc; ++j) {
    pos[b * Sc + LTc + j] = (float)(sp + j);
    am[b * Sc + LTc + j] = (mm[b * Tc + (j - 1)] > 0.f) ? 1 : 0;
  }
}

__global__ void k_build_x(const int* __restrict__ ids, const float* __restrict__ embed,
                          const float* __restrict__ start, const float* __restrict__ encm,
                          float* __restrict__ x) {
  size_t i = (size_t)blockIdx.x * 256 + threadIdx.x;
  if (i >= (size_t)MS * Hc) return;
  int h = (int)(i % Hc); size_t bs = i / Hc;
  int b = (int)(bs / Sc), s = (int)(bs % Sc);
  float v;
  if (s < LTc)       v = embed[(size_t)ids[b * LTc + s] * Hc + h];
  else if (s == LTc) v = start[h];
  else               v = encm[((size_t)b * Tc + (s - LTc - 1)) * Hc + h];
  x[i] = v;
}

// ---------------- RoPE (in place on q or k) ----------------
__global__ void k_rope(float* __restrict__ x, const float* __restrict__ pos) {
  size_t i = (size_t)blockIdx.x * 256 + threadIdx.x;
  const size_t tot = (size_t)MS * NHc * (DHc / 2);
  if (i >= tot) return;
  int f = (int)(i % (DHc / 2)); size_t r = i / (DHc / 2);
  int hh = (int)(r % NHc); size_t bs = r / NHc;
  float p = pos[bs];
  float inv = powf(10000.f, -(float)f / (float)(DHc / 2));
  float ang = p * inv;
  float c = cosf(ang), sn = sinf(ang);
  float* base = x + (bs * NHc + hh) * DHc;
  float x1 = base[f], x2 = base[f + DHc / 2];
  base[f]           = x1 * c - x2 * sn;
  base[f + DHc / 2] = x2 * c + x1 * sn;
}

// ---------------- attention: one block per (b,h); K/V staged once in LDS ----
// LDS: K,V as bf16 (2*324*66*2B ~ 85KB) << 320KB WGP LDS.
// 256 threads = 4 query rows x 64 "key/dim lanes" per sweep.
#define ATQ 4
__global__ __launch_bounds__(256)
void k_attn(const float* __restrict__ q, const float* __restrict__ k,
            const float* __restrict__ v, const int* __restrict__ am,
            float* __restrict__ o) {
  const int h = blockIdx.x & (NHc - 1);
  const int b = blockIdx.x >> 4;
  __shared__ __bf16 Kb[Sc][DHc + 2];
  __shared__ __bf16 Vb[Sc][DHc + 2];
  __shared__ float qs[ATQ][DHc];
  __shared__ float sc[ATQ][Sc];
  __shared__ float rbuf[256];
  const int tid = threadIdx.x;
  const int ql  = tid >> 6;   // query slot 0..3
  const int ln  = tid & 63;   // lane within query group

  for (int i = tid; i < Sc * DHc; i += 256) {
    int ks = i / DHc, d = i % DHc;
    size_t base = (((size_t)b * Sc + ks) * NHc + h) * DHc + d;
    Kb[ks][d] = (__bf16)k[base];
    Vb[ks][d] = (__bf16)v[base];
  }
  __syncthreads();

  for (int q0 = 0; q0 < Sc; q0 += ATQ) {     // 324/4 = 81 sweeps
    const int qrow = q0 + ql;
    // stage 4 query rows (one element per thread)
    qs[ql][ln] = q[(((size_t)b * Sc + qrow) * NHc + h) * DHc + ln];
    __syncthreads();

    // scores for this query row (keys strided across 64 threads)
    float lmax = -1e30f;
    for (int ks = ln; ks < Sc; ks += 64) {
      float dsum = 0.f;
      #pragma unroll
      for (int d = 0; d < DHc; ++d) dsum += qs[ql][d] * (float)Kb[ks][d];
      dsum *= 0.125f;
      bool ok = (ks <= qrow) && (am[b * Sc + ks] != 0);
      dsum = ok ? dsum : -1e9f;
      sc[ql][ks] = dsum;
      lmax = fmaxf(lmax, dsum);
    }
    rbuf[tid] = lmax; __syncthreads();
    for (int st = 32; st > 0; st >>= 1) {
      if (ln < st) rbuf[tid] = fmaxf(rbuf[tid], rbuf[tid + st]);
      __syncthreads();
    }
    float mx = rbuf[ql * 64]; __syncthreads();

    float lsum = 0.f;
    for (int ks = ln; ks < Sc; ks += 64) {
      float e = expf(sc[ql][ks] - mx); sc[ql][ks] = e; lsum += e;
    }
    rbuf[tid] = lsum; __syncthreads();
    for (int st = 32; st > 0; st >>= 1) {
      if (ln < st) rbuf[tid] += rbuf[tid + st];
      __syncthreads();
    }
    float inv = 1.f / rbuf[ql * 64];
    __syncthreads();

    // output: thread = (query slot, head dim), reduce over keys from LDS
    float acc = 0.f;
    for (int ks = 0; ks <= qrow; ++ks) acc += sc[ql][ks] * (float)Vb[ks][ln];
    o[(((size_t)b * Sc + qrow) * NHc + h) * DHc + ln] = acc * inv;
    __syncthreads();
  }
}

// ---------------- SwiGLU combine ----------------
__global__ void k_silu_mul(const float* __restrict__ g, const float* __restrict__ u,
                           __bf16* __restrict__ out, size_t n) {
  size_t i = (size_t)blockIdx.x * 256 + threadIdx.x;
  if (i >= n) return;
  float x = g[i];
  out[i] = (__bf16)((x / (1.f + expf(-x))) * u[i]);
}

// ---------------- gathers ----------------
__global__ void k_gather_mh(const __bf16* __restrict__ src, __bf16* __restrict__ dst) {
  size_t i = (size_t)blockIdx.x * 256 + threadIdx.x;
  if (i >= (size_t)MT * Hc) return;
  int h = (int)(i % Hc); size_t r = i / Hc;
  int b = (int)(r / Tc), t = (int)(r % Tc);
  dst[i] = src[((size_t)b * Sc + LTc + t) * Hc + h];
}
__global__ void k_gather_lm(const __bf16* __restrict__ src, __bf16* __restrict__ dst) {
  size_t i = (size_t)blockIdx.x * 256 + threadIdx.x;
  if (i >= (size_t)MLM * Hc) return;
  int h = (int)(i % Hc); size_t r = i / Hc;
  int b = (int)(r / (LTc - 1)), t = (int)(r % (LTc - 1));
  dst[i] = src[((size_t)b * Sc + t) * Hc + h];
}

// ---------------- online log-softmax over vocab chunks ----------------
__global__ void k_lse_init(float* rmax, float* rsum, float* rtgt) {
  int i = blockIdx.x * 256 + threadIdx.x;
  if (i >= MLM) return;
  rmax[i] = -1e30f; rsum[i] = 0.f; rtgt[i] = 0.f;
}
__global__ void k_lse_update(const float* __restrict__ logits, int v0,
                             const int* __restrict__ ids,
                             float* rmax, float* rsum, float* rtgt) {
  int row = blockIdx.x;  // 0..MLM-1
  int tid = threadIdx.x;
  __shared__ float red[256];
  const float* lp = logits + (size_t)row * LCH;
  float lm = -1e30f;
  for (int j = tid; j < LCH; j += 256) lm = fmaxf(lm, lp[j]);
  red[tid] = lm; __syncthreads();
  for (int st = 128; st > 0; st >>= 1) { if (tid < st) red[tid] = fmaxf(red[tid], red[tid + st]); __syncthreads(); }
  float cmax = red[0]; __syncthreads();
  float om = rmax[row];
  float nm = fmaxf(om, cmax);
  float ls = 0.f;
  for (int j = tid; j < LCH; j += 256) ls += expf(lp[j] - nm);
  red[tid] = ls; __syncthreads();
  for (int st = 128; st > 0; st >>= 1) { if (tid < st) red[tid] += red[tid + st]; __syncthreads(); }
  if (tid == 0) {
    float s = rsum[row] * expf(om - nm) + red[0];
    rsum[row] = s; rmax[row] = nm;
    int b = row / (LTc - 1), i2 = row % (LTc - 1);
    int label = ids[b * LTc + i2 + 1];
    if (label >= v0 && label < v0 + LCH) rtgt[row] = lp[label - v0];
  }
}

// ---------------- losses ----------------
__global__ void k_zero_acc(float* acc) {
  int i = threadIdx.x; if (i < 8) acc[i] = 0.f;
}
__global__ void k_loss_pv(const float* __restrict__ pred, const float* __restrict__ motion,
                          const float* __restrict__ mm, float* acc) {
  int i = blockIdx.x * 256 + threadIdx.x;
  if (i >= Bc * Tc) return;
  int t = i % Tc;
  const float* p = pred + (size_t)i * Dc;
  const float* m = motion + (size_t)i * Dc;
  float s = 0.f;
  for (int d = 0; d < Dc; ++d) { float df = p[d] - m[d]; s += df * df; }
  s /= (float)Dc;
  float msk = mm[i];
  atomicAdd(&acc[0], s * msk);
  atomicAdd(&acc[1], msk);
  if (t > 0) {
    const float* p0 = p - Dc; const float* m0 = m - Dc;
    float sv = 0.f;
    for (int d = 0; d < Dc; ++d) {
      float dv = (p[d] - p0[d]) - (m[d] - m0[d]); sv += dv * dv;
    }
    sv /= (float)Dc;
    atomicAdd(&acc[2], sv * msk);
    atomicAdd(&acc[3], msk);
  }
}
__global__ void k_loss_lang(const float* rmax, const float* rsum, const float* rtgt,
                            const int* __restrict__ ids, float* acc) {
  int row = blockIdx.x * 256 + threadIdx.x;
  if (row >= MLM) return;
  int b = row / (LTc - 1), i2 = row % (LTc - 1);
  int label = ids[b * LTc + i2 + 1];
  float valid = (label != 0) ? 1.f : 0.f;
  float nll = -(rtgt[row] - rmax[row] - logf(rsum[row]));
  atomicAdd(&acc[4], nll * valid);
  atomicAdd(&acc[5], valid);
}
__global__ void k_finalize(const float* acc, float* out) {
  if (threadIdx.x == 0)
    out[0] = acc[0] / (acc[1] + 1e-8f) + acc[2] / (acc[3] + 1e-8f)
           + acc[4] / fmaxf(acc[5], 1.0f);
}

// =======================================================================
static inline unsigned cdiv(size_t a, size_t b) { return (unsigned)((a + b - 1) / b); }

extern "C" void kernel_launch(void* const* d_in, const int* in_sizes, int n_in,
                              void* d_out, int out_size, void* d_ws, size_t ws_size,
                              hipStream_t stream) {
  (void)in_sizes; (void)n_in; (void)out_size; (void)ws_size;
  const int*   ids      = (const int*)  d_in[0];
  const float* motion   = (const float*)d_in[1];
  const float* mmask    = (const float*)d_in[2];
  const float* embed    = (const float*)d_in[3];
  const float* lm_head  = (const float*)d_in[4];
  const float* ln1      = (const float*)d_in[5];
  const float* ln2      = (const float*)d_in[6];
  const float* fnorm    = (const float*)d_in[7];
  const float* wq       = (const float*)d_in[8];
  const float* wk       = (const float*)d_in[9];
  const float* wv       = (const float*)d_in[10];
  const float* wo       = (const float*)d_in[11];
  const float* wg       = (const float*)d_in[12];
  const float* wu       = (const float*)d_in[13];
  const float* wd       = (const float*)d_in[14];
  const float* start    = (const float*)d_in[15];
  const float* enc_w1   = (const float*)d_in[16];
  const float* enc_b1   = (const float*)d_in[17];
  const float* enc_w2   = (const float*)d_in[18];
  const float* enc_b2   = (const float*)d_in[19];
  const float* enc_w3   = (const float*)d_in[20];
  const float* enc_b3   = (const float*)d_in[21];
  const float* enc_g    = (const float*)d_in[22];
  const float* enc_bb   = (const float*)d_in[23];
  const float* dec_w1   = (const float*)d_in[24];
  const float* dec_b1   = (const float*)d_in[25];
  const float* dec_w2   = (const float*)d_in[26];
  const float* dec_b2   = (const float*)d_in[27];
  const float* dec_w3   = (const float*)d_in[28];
  const float* dec_b3   = (const float*)d_in[29];

  float* outv  = (float*)d_out;
  float* predO = outv + 1;               // (B,T,D) flat

  // ---- workspace carve-out ----
  size_t off = 0;
  auto alloc = [&](size_t bytes) -> void* {
    off = (off + 255) & ~(size_t)255;
    void* p = (char*)d_ws + off;
    off += bytes;
    return p;
  };
  __bf16* enc1tr = (__bf16*)alloc((size_t)Hc * DPAD * 2);
  __bf16* enc2tr = (__bf16*)alloc((size_t)Hc * Hc * 2);
  __bf16* enc3tr = (__bf16*)alloc((size_t)Hc * Hc * 2);
  __bf16* dec1tr = (__bf16*)alloc((size_t)Hc * Hc * 2);
  __bf16* dec2tr = (__bf16*)alloc((size_t)Hc * Hc * 2);
  __bf16* dec3tr = (__bf16*)alloc((size_t)Dc * Hc * 2);
  __bf16* wqtr   = (__bf16*)alloc((size_t)NLc * Hc * Hc * 2);
  __bf16* wktr   = (__bf16*)alloc((size_t)NLc * Hc * Hc * 2);
  __bf16* wvtr   = (__bf16*)alloc((size_t)NLc * Hc * Hc * 2);
  __bf16* wotr   = (__bf16*)alloc((size_t)NLc * Hc * Hc * 2);
  __bf16* wgtr   = (__bf16*)alloc((size_t)NLc * Fc * Hc * 2);
  __bf16* wutr   = (__bf16*)alloc((size_t)NLc * Fc * Hc * 2);
  __bf16* wdtr   = (__bf16*)alloc((size_t)NLc * Hc * Fc * 2);
  __bf16* lmtr   = (__bf16*)alloc((size_t)Vc * Hc * 2);
  __bf16* mbf    = (__bf16*)alloc((size_t)MT * DPAD * 2);
  __bf16* encbf  = (__bf16*)alloc((size_t)MT * Hc * 2);
  __bf16* xnbf   = (__bf16*)alloc((size_t)MS * Hc * 2);
  __bf16* obf    = (__bf16*)alloc((size_t)MS * Hc * 2);
  __bf16* ffbf   = (__bf16*)alloc((size_t)MS * Fc * 2);
  __bf16* hnbf   = (__bf16*)alloc((size_t)MS * Hc * 2);
  __bf16* mhbf   = (__bf16*)alloc((size_t)MT * Hc * 2);
  __bf16* dbf    = (__bf16*)alloc((size_t)MT * Hc * 2);
  __bf16* lmA    = (__bf16*)alloc((size_t)MLM * Hc * 2);
  float*  e1     = (float*)alloc((size_t)MT * Hc * 4);
  float*  e2     = (float*)alloc((size_t)MT * Hc * 4);
  float*  xb     = (float*)alloc((size_t)MS * Hc * 4);
  float*  qb     = (float*)alloc((size_t)MS * Hc * 4);
  float*  kb     = (float*)alloc((size_t)MS * Hc * 4);
  float*  vb     = (float*)alloc((size_t)MS * Hc * 4);
  float*  ob     = (float*)alloc((size_t)MS * Hc * 4);
  float*  ffg    = (float*)alloc((size_t)MS * Fc * 4);
  float*  ffu    = (float*)alloc((size_t)MS * Fc * 4);
  float*  d1     = (float*)alloc((size_t)MT * Hc * 4);
  float*  d2     = (float*)alloc((size_t)MT * Hc * 4);
  float*  logits = (float*)alloc((size_t)MLM * LCH * 4);
  float*  posb   = (float*)alloc((size_t)Bc * Sc * 4);
  int*    amb    = (int*)  alloc((size_t)Bc * Sc * 4);
  float*  rmax   = (float*)alloc((size_t)MLM * 4);
  float*  rsum   = (float*)alloc((size_t)MLM * 4);
  float*  rtgt   = (float*)alloc((size_t)MLM * 4);
  float*  accb   = (float*)alloc(8 * 4);

  auto cvt_tr = [&](const float* s, __bf16* dst, int K, int N, int Kpad) {
    k_cvt_tr<<<cdiv((size_t)N * Kpad, 256), 256, 0, stream>>>(s, dst, K, N, Kpad);
  };
  auto cvt = [&](const float* s, __bf16* dst, int rows, int K, int Kpad) {
    k_cvt_pad<<<cdiv((size_t)rows * Kpad, 256), 256, 0, stream>>>(s, dst, rows, K, Kpad);
  };
  auto gemm = [&](const __bf16* A, const __bf16* B, float* C, const float* bias,
                  const float* res, int M, int N, int K, int op) {
    dim3 g(cdiv(M, TMt), cdiv(N, TNt));
    k_gemm<<<g, 256, 0, stream>>>(A, B, C, bias, res, M, N, K, op);
  };

  // ---- 1. weight conversion (fp32 -> bf16, [N][K] transposed) ----
  cvt_tr(enc_w1, enc1tr, Dc, Hc, DPAD);
  cvt_tr(enc_w2, enc2tr, Hc, Hc, Hc);
  cvt_tr(enc_w3, enc3tr, Hc, Hc, Hc);
  cvt_tr(dec_w1, dec1tr, Hc, Hc, Hc);
  cvt_tr(dec_w2, dec2tr, Hc, Hc, Hc);
  cvt_tr(dec_w3, dec3tr, Hc, Dc, Hc);
  for (int l = 0; l < NLc; ++l) {
    cvt_tr(wq + (size_t)l * Hc * Hc, wqtr + (size_t)l * Hc * Hc, Hc, Hc, Hc);
    cvt_tr(wk + (size_t)l * Hc * Hc, wktr + (size_t)l * Hc * Hc, Hc, Hc, Hc);
    cvt_tr(wv + (size_t)l * Hc * Hc, wvtr + (size_t)l * Hc * Hc, Hc, Hc, Hc);
    cvt_tr(wo + (size_t)l * Hc * Hc, wotr + (size_t)l * Hc * Hc, Hc, Hc, Hc);
    cvt_tr(wg + (size_t)l * Hc * Fc, wgtr + (size_t)l * Fc * Hc, Hc, Fc, Hc);
    cvt_tr(wu + (size_t)l * Hc * Fc, wutr + (size_t)l * Fc * Hc, Hc, Fc, Hc);
    cvt_tr(wd + (size_t)l * Fc * Hc, wdtr + (size_t)l * Hc * Fc, Fc, Hc, Fc);
  }
  cvt_tr(lm_head, lmtr, Hc, Vc, Hc);

  // ---- 2. motion encoder ----
  cvt(motion, mbf, MT, Dc, DPAD);
  gemm(mbf, enc1tr, e1, enc_b1, nullptr, MT, Hc, DPAD, 2);
  cvt(e1, encbf, MT, Hc, Hc);
  gemm(encbf, enc2tr, e2, enc_b2, nullptr, MT, Hc, Hc, 2);
  cvt(e2, encbf, MT, Hc, Hc);
  gemm(encbf, enc3tr, e1, enc_b3, nullptr, MT, Hc, Hc, 1);
  k_layernorm<<<MT, 256, 0, stream>>>(e1, enc_g, enc_bb, e2, MT);

  // ---- 3. sequence assembly ----
  k_posmask<<<Bc, 32, 0, stream>>>(ids, mmask, posb, amb);
  k_build_x<<<cdiv((size_t)MS * Hc, 256), 256, 0, stream>>>(ids, embed, start, e2, xb);

  // ---- 4. transformer layers ----
  for (int l = 0; l < NLc; ++l) {
    k_rmsnorm<<<MS, 256, 0, stream>>>(xb, ln1 + (size_t)l * Hc, xnbf, MS);
    gemm(xnbf, wqtr + (size_t)l * Hc * Hc, qb, nullptr, nullptr, MS, Hc, Hc, 0);
    gemm(xnbf, wktr + (size_t)l * Hc * Hc, kb, nullptr, nullptr, MS, Hc, Hc, 0);
    gemm(xnbf, wvtr + (size_t)l * Hc * Hc, vb, nullptr, nullptr, MS, Hc, Hc, 0);
    unsigned rg = cdiv((size_t)MS * NHc * (DHc / 2), 256);
    k_rope<<<rg, 256, 0, stream>>>(qb, posb);
    k_rope<<<rg, 256, 0, stream>>>(kb, posb);
    k_attn<<<Bc * NHc, 256, 0, stream>>>(qb, kb, vb, amb, ob);
    cvt(ob, obf, MS, Hc, Hc);
    gemm(obf, wotr + (size_t)l * Hc * Hc, xb, nullptr, xb, MS, Hc, Hc, 3);
    k_rmsnorm<<<MS, 256, 0, stream>>>(xb, ln2 + (size_t)l * Hc, xnbf, MS);
    gemm(xnbf, wgtr + (size_t)l * Fc * Hc, ffg, nullptr, nullptr, MS, Fc, Hc, 0);
    gemm(xnbf, wutr + (size_t)l * Fc * Hc, ffu, nullptr, nullptr, MS, Fc, Hc, 0);
    k_silu_mul<<<cdiv((size_t)MS * Fc, 256), 256, 0, stream>>>(ffg, ffu, ffbf, (size_t)MS * Fc);
    gemm(ffbf, wdtr + (size_t)l * Hc * Fc, xb, nullptr, xb, MS, Hc, Fc, 3);
  }

  // ---- 5. final norm + motion decoder -> pred ----
  k_rmsnorm<<<MS, 256, 0, stream>>>(xb, fnorm, hnbf, MS);
  k_gather_mh<<<cdiv((size_t)MT * Hc, 256), 256, 0, stream>>>(hnbf, mhbf);
  gemm(mhbf, dec1tr, d1, dec_b1, nullptr, MT, Hc, Hc, 2);
  cvt(d1, dbf, MT, Hc, Hc);
  gemm(dbf, dec2tr, d2, dec_b2, nullptr, MT, Hc, Hc, 2);
  cvt(d2, dbf, MT, Hc, Hc);
  gemm(dbf, dec3tr, predO, dec_b3, nullptr, MT, Dc, Hc, 1);

  // ---- 6. LM head with online log-softmax over vocab chunks ----
  k_gather_lm<<<cdiv((size_t)MLM * Hc, 256), 256, 0, stream>>>(hnbf, lmA);
  k_lse_init<<<cdiv(MLM, 256), 256, 0, stream>>>(rmax, rsum, rtgt);
  for (int ch = 0; ch < Vc / LCH; ++ch) {
    gemm(lmA, lmtr + (size_t)ch * LCH * Hc, logits, nullptr, nullptr, MLM, LCH, Hc, 0);
    k_lse_update<<<MLM, 256, 0, stream>>>(logits, ch * LCH, ids, rmax, rsum, rtgt);
  }

  // ---- 7. losses ----
  k_zero_acc<<<1, 32, 0, stream>>>(accb);
  k_loss_pv<<<cdiv(Bc * Tc, 256), 256, 0, stream>>>(predO, motion, mmask, accb);
  k_loss_lang<<<cdiv(MLM, 256), 256, 0, stream>>>(rmax, rsum, rtgt, ids, accb);
  k_finalize<<<1, 32, 0, stream>>>(accb, outv);
}
